// STMamba_5712306503726
// MI455X (gfx1250) — compile-verified, matrix operand order checked by reading, not run on hardware
//
#include <hip/hip_runtime.h>
#include <math.h>

typedef __bf16 bf16_t;
typedef __attribute__((ext_vector_type(16))) __bf16 v16bf;
typedef __attribute__((ext_vector_type(8)))  float  v8f;

#define NTOK   32768        // Bv * T*H*W = 4 * 8192
#define LSEQ   8192
#define DIMC   96
#define DINNER 192
#define DSTATE 16
#define DTRANK 6
#define HIDC   384
#define TT     8
#define HH     32
#define WW     32

__device__ __forceinline__ float siluf(float x)  { return x / (1.f + __expf(-x)); }
__device__ __forceinline__ float softplusf(float x) { return (x > 20.f) ? x : log1pf(__expf(x)); }

// LDS byte offset (wave-relative) of a pointer into __shared__ memory.
__device__ __forceinline__ unsigned lds_addr_u32(const void* p) {
    return (unsigned)(uintptr_t)(const __attribute__((address_space(3))) void*)p;
}

// ---------------------------------------------------------------- convert
__global__ void f32_to_bf16_kernel(const float* __restrict__ in, bf16_t* __restrict__ out, int n) {
    int i = blockIdx.x * blockDim.x + threadIdx.x;
    if (i < n) out[i] = (bf16_t)in[i];
}

// ---------------------------------------------------------------- LN1 + layout transpose
// x_in: (Bv*T, C, H, W). token m = bv*8192 + t*1024 + h*32 + w. One wave per token.
__global__ void ln1_prep_kernel(const float* __restrict__ x_in,
                                const float* __restrict__ w, const float* __restrict__ b,
                                float* __restrict__ x_flat, bf16_t* __restrict__ xn_bf) {
    int token = blockIdx.x * (blockDim.x >> 5) + (threadIdx.x >> 5);
    int lane  = threadIdx.x & 31;
    if (token >= NTOK) return;
    int bv = token >> 13, r = token & (LSEQ - 1);
    int t = r >> 10, hw = r & 1023;
    float vals[3];
    #pragma unroll
    for (int i = 0; i < 3; ++i) {
        int c = lane + 32 * i;
        vals[i] = x_in[(((size_t)(bv * TT + t) * DIMC) + c) * 1024 + hw];
    }
    float s = vals[0] + vals[1] + vals[2];
    #pragma unroll
    for (int m = 16; m >= 1; m >>= 1) s += __shfl_xor(s, m, 32);
    float mu = s * (1.f / 96.f);
    float q = 0.f;
    #pragma unroll
    for (int i = 0; i < 3; ++i) { float d = vals[i] - mu; q += d * d; }
    #pragma unroll
    for (int m = 16; m >= 1; m >>= 1) q += __shfl_xor(q, m, 32);
    float rstd = rsqrtf(q * (1.f / 96.f) + 1e-5f);
    #pragma unroll
    for (int i = 0; i < 3; ++i) {
        int c = lane + 32 * i;
        float xf = vals[i];
        x_flat[(size_t)token * DIMC + c] = xf;
        xn_bf[(size_t)token * DIMC + c] = (bf16_t)((xf - mu) * rstd * w[c] + b[c]);
    }
}

// ---------------------------------------------------------------- LN2 (token-major f32 in)
__global__ void ln2_kernel(const float* __restrict__ xin,
                           const float* __restrict__ w, const float* __restrict__ b,
                           bf16_t* __restrict__ out_bf) {
    int token = blockIdx.x * (blockDim.x >> 5) + (threadIdx.x >> 5);
    int lane  = threadIdx.x & 31;
    if (token >= NTOK) return;
    float vals[3];
    #pragma unroll
    for (int i = 0; i < 3; ++i) vals[i] = xin[(size_t)token * DIMC + lane + 32 * i];
    float s = vals[0] + vals[1] + vals[2];
    #pragma unroll
    for (int m = 16; m >= 1; m >>= 1) s += __shfl_xor(s, m, 32);
    float mu = s * (1.f / 96.f);
    float q = 0.f;
    #pragma unroll
    for (int i = 0; i < 3; ++i) { float d = vals[i] - mu; q += d * d; }
    #pragma unroll
    for (int m = 16; m >= 1; m >>= 1) q += __shfl_xor(q, m, 32);
    float rstd = rsqrtf(q * (1.f / 96.f) + 1e-5f);
    #pragma unroll
    for (int i = 0; i < 3; ++i) {
        int c = lane + 32 * i;
        out_bf[(size_t)token * DIMC + c] = (bf16_t)((vals[i] - mu) * rstd * w[c] + b[c]);
    }
}

// ---------------------------------------------------------------- WMMA GEMM
// C[M,N] = A[M,K] @ W[N,K]^T (+bias[n]) (+resid).
// Block = one 16-row M-tile; A tile (16 x K, contiguous) is DMA'd to LDS with
// global_load_async_to_lds_b128. Each wave owns NB consecutive 16-col N-tiles,
// reusing one A fragment for NB WMMAs per K-step. blockDim.x = (N/16/NB)*32.
// mode 0: float out row-major (optional resid). mode 1: bf16 out.
// mode 2: float out in (Bv*T, C, H, W) video layout with resid (N must be 96).
template <int NB>
__global__ void wmma_gemm_kernel(const bf16_t* __restrict__ A, const bf16_t* __restrict__ W,
                                 const float* __restrict__ bias, const float* __restrict__ resid,
                                 float* __restrict__ outf, bf16_t* __restrict__ outbf,
                                 int M, int N, int K, int mode) {
    __shared__ __align__(16) bf16_t lds_a[16 * 384];   // up to K=384
    const int tid  = threadIdx.x;
    const int lane = tid & 31;
    const int wv   = tid >> 5;
    const int tm   = blockIdx.x;

    // ---- async DMA of the contiguous 16xK A tile into LDS ----
    {
        const char* gsrc  = (const char*)(A + (size_t)tm * 16 * K);
        unsigned    lbase = lds_addr_u32(lds_a);
        const int nbytes  = 16 * K * 2;
        for (int ofs = tid * 16; ofs < nbytes; ofs += blockDim.x * 16) {
            unsigned           ldst = lbase + ofs;
            unsigned long long ga   = (unsigned long long)(uintptr_t)(gsrc + ofs);
            asm volatile("global_load_async_to_lds_b128 %0, %1, off"
                         :: "v"(ldst), "v"(ga) : "memory");
        }
        asm volatile("s_wait_asynccnt 0" ::: "memory");
        __syncthreads();
    }

    const int rowl   = lane & 15;          // local A row / output col-in-tile
    const int khalf  = (lane >> 4) * 8;
    const int base_tn = wv * NB;

    v8f acc[NB];
    #pragma unroll
    for (int j = 0; j < NB; ++j) acc[j] = (v8f){0.f, 0.f, 0.f, 0.f, 0.f, 0.f, 0.f, 0.f};

    for (int k0 = 0; k0 < K; k0 += 32) {
        v16bf af;
        #pragma unroll
        for (int v = 0; v < 8; ++v) {
            // 16-bit A/B fragment K mapping (ISA 7.12.2): lanes<16 -> K 0..7,16..23 ; lanes>=16 -> +8
            int kb = ((v < 4) ? (2 * v) : (16 + 2 * (v - 4))) + khalf + k0;
            af[2 * v]     = lds_a[rowl * K + kb];
            af[2 * v + 1] = lds_a[rowl * K + kb + 1];
        }
        #pragma unroll
        for (int j = 0; j < NB; ++j) {
            size_t col = (size_t)((base_tn + j) * 16 + rowl) * K;
            v16bf bfr;
            #pragma unroll
            for (int v = 0; v < 8; ++v) {
                int kb = ((v < 4) ? (2 * v) : (16 + 2 * (v - 4))) + khalf + k0;
                bfr[2 * v]     = W[col + kb];
                bfr[2 * v + 1] = W[col + kb + 1];
            }
            acc[j] = __builtin_amdgcn_wmma_f32_16x16x32_bf16(false, af, false, bfr,
                                                             (short)0, acc[j], false, false);
        }
    }

    const int nl = lane & 15;
    const int mbase = (lane >> 4) * 8;
    #pragma unroll
    for (int j = 0; j < NB; ++j) {
        #pragma unroll
        for (int v = 0; v < 8; ++v) {
            int m = tm * 16 + mbase + v;
            int n = (base_tn + j) * 16 + nl;
            float val = acc[j][v];
            if (bias)  val += bias[n];
            if (mode == 0) {
                if (resid) val += resid[(size_t)m * N + n];
                outf[(size_t)m * N + n] = val;
            } else if (mode == 1) {
                outbf[(size_t)m * N + n] = (bf16_t)val;
            } else {
                val += resid[(size_t)m * N + n];
                int bv = m >> 13, r = m & (LSEQ - 1), t = r >> 10, hw = r & 1023;
                outf[(((size_t)(bv * TT + t) * DIMC) + n) * 1024 + hw] = val;
            }
        }
    }
}

// ---------------------------------------------------------------- causal depthwise conv4 + SiLU
// xm lives in xz (stride 384, cols 0..191). reverse=1 runs on the time-reversed sequence.
__global__ void conv_silu_kernel(const float* __restrict__ xz, const float* __restrict__ w,
                                 const float* __restrict__ bias, float* __restrict__ u,
                                 int reverse) {
    int idx = blockIdx.x * blockDim.x + threadIdx.x;
    if (idx >= NTOK * DINNER) return;
    int d = idx % DINNER, bl = idx / DINNER;
    int b = bl >> 13, l = bl & (LSEQ - 1);
    float acc = bias[d];
    #pragma unroll
    for (int k = 0; k < 4; ++k) {
        int ls = l - 3 + k;
        if (ls >= 0) {
            int lsrc = reverse ? (LSEQ - 1 - ls) : ls;
            acc += w[d * 4 + k] * xz[((size_t)(b * LSEQ + lsrc)) * (2 * DINNER) + d];
        }
    }
    u[(size_t)bl * DINNER + d] = siluf(acc);
}

// ---------------------------------------------------------------- x_proj (192 -> 38), scalar
__global__ void xproj_kernel(const float* __restrict__ u, const float* __restrict__ xw,
                             float* __restrict__ dtBC) {
    int idx = blockIdx.x * blockDim.x + threadIdx.x;
    if (idx >= NTOK * (DTRANK + 2 * DSTATE)) return;
    int k = idx % 38, bl = idx / 38;
    const float* up = u  + (size_t)bl * DINNER;
    const float* wp = xw + (size_t)k * DINNER;
    float s = 0.f;
    #pragma unroll 8
    for (int d = 0; d < DINNER; ++d) s += up[d] * wp[d];
    dtBC[idx] = s;
}

// ---------------------------------------------------------------- dt_proj + softplus
__global__ void delta_kernel(const float* __restrict__ dtBC, const float* __restrict__ dtw,
                             const float* __restrict__ dtb, float* __restrict__ delta) {
    int idx = blockIdx.x * blockDim.x + threadIdx.x;
    if (idx >= NTOK * DINNER) return;
    int d = idx % DINNER, bl = idx / DINNER;
    float s = dtb[d];
    #pragma unroll
    for (int r = 0; r < DTRANK; ++r) s += dtBC[(size_t)bl * 38 + r] * dtw[d * DTRANK + r];
    delta[idx] = softplusf(s);
}

// ---------------------------------------------------------------- selective scan
// 2 (b,d) channels per wave: lanes 0-15 states of ch0, lanes 16-31 states of ch1.
__global__ void scan_kernel(const float* __restrict__ u, const float* __restrict__ delta,
                            const float* __restrict__ dtBC, const float* __restrict__ A_log,
                            const float* __restrict__ Dv, float* __restrict__ y) {
    int gid  = blockIdx.x * blockDim.x + threadIdx.x;
    int lane = threadIdx.x & 31;
    int ch   = (gid >> 5) * 2 + (lane >> 4);
    if (ch >= 4 * DINNER) return;
    int b = ch / DINNER, d = ch % DINNER, n = lane & 15;
    float A  = -__expf(A_log[d * DSTATE + n]);
    float Dd = Dv[d];
    float h = 0.f;
    size_t base_ud = (size_t)b * LSEQ * DINNER + d;
    size_t base_bc = (size_t)b * LSEQ * 38;
    #pragma unroll 2
    for (int l = 0; l < LSEQ; ++l) {
        size_t o = base_ud + (size_t)l * DINNER;
        if (l + 16 < LSEQ) {
            __builtin_prefetch(&delta[o + 16 * DINNER], 0, 1);
            __builtin_prefetch(&u[o + 16 * DINNER], 0, 1);
        }
        float dl = delta[o];
        float ul = u[o];
        float Bc = dtBC[base_bc + (size_t)l * 38 + DTRANK + n];
        float Cc = dtBC[base_bc + (size_t)l * 38 + DTRANK + DSTATE + n];
        h = h * __expf(dl * A) + dl * ul * Bc;
        float rsum = h * Cc;
        rsum += __shfl_xor(rsum, 1, 32);
        rsum += __shfl_xor(rsum, 2, 32);
        rsum += __shfl_xor(rsum, 4, 32);
        rsum += __shfl_xor(rsum, 8, 32);
        if (n == 0) y[o] = rsum + ul * Dd;
    }
}

// ---------------------------------------------------------------- gate + fwd/bwd combine
__global__ void gate_combine_kernel(const float* __restrict__ xz, const float* __restrict__ y_f,
                                    const float* __restrict__ y_b, bf16_t* __restrict__ yc_bf) {
    int idx = blockIdx.x * blockDim.x + threadIdx.x;
    if (idx >= NTOK * DINNER) return;
    int d = idx % DINNER, bl = idx / DINNER;
    int b = bl >> 13, l = bl & (LSEQ - 1);
    float z = xz[(size_t)bl * (2 * DINNER) + DINNER + d];
    float yb = y_b[(size_t)(b * LSEQ + (LSEQ - 1 - l)) * DINNER + d];
    yc_bf[idx] = (bf16_t)((y_f[idx] + yb) * siluf(z));
}

// ---------------------------------------------------------------- depthwise 3x3x3 + GELU
__global__ void dwconv3d_gelu_kernel(const bf16_t* __restrict__ in, const float* __restrict__ w,
                                     const float* __restrict__ bias, bf16_t* __restrict__ out,
                                     int C) {
    int idx = blockIdx.x * blockDim.x + threadIdx.x;
    if (idx >= NTOK * C) return;
    int c = idx % C, m = idx / C;
    int bv = m >> 13, r = m & (LSEQ - 1);
    int t = r >> 10, h = (r >> 5) & 31, x = r & 31;
    float acc = 0.f;
    #pragma unroll
    for (int dt = -1; dt <= 1; ++dt)
        #pragma unroll
        for (int dh = -1; dh <= 1; ++dh)
            #pragma unroll
            for (int dw = -1; dw <= 1; ++dw) {
                int tt = t + dt, h2 = h + dh, w2 = x + dw;
                if (tt >= 0 && tt < TT && h2 >= 0 && h2 < HH && w2 >= 0 && w2 < WW) {
                    int mm = (bv << 13) + (tt << 10) + (h2 << 5) + w2;
                    acc += (float)in[(size_t)mm * C + c] *
                           w[c * 27 + (dt + 1) * 9 + (dh + 1) * 3 + (dw + 1)];
                }
            }
    acc += bias[c];
    out[idx] = (bf16_t)(0.5f * acc * (1.f + erff(acc * 0.70710678f)));
}

// ================================================================ host
extern "C" void kernel_launch(void* const* d_in, const int* in_sizes, int n_in,
                              void* d_out, int out_size, void* d_ws, size_t ws_size,
                              hipStream_t stream) {
    const float* x_in      = (const float*)d_in[0];
    const float* ln1_w     = (const float*)d_in[1];
    const float* ln1_b     = (const float*)d_in[2];
    const float* ln2_w     = (const float*)d_in[3];
    const float* ln2_b     = (const float*)d_in[4];
    const float* in_proj_w = (const float*)d_in[5];
    const float* conv_f_w  = (const float*)d_in[6];
    const float* conv_f_b  = (const float*)d_in[7];
    const float* xproj_f_w = (const float*)d_in[8];
    const float* dtprj_f_w = (const float*)d_in[9];
    const float* dtprj_f_b = (const float*)d_in[10];
    const float* A_f_log   = (const float*)d_in[11];
    const float* D_f       = (const float*)d_in[12];
    const float* conv_b_w  = (const float*)d_in[13];
    const float* conv_b_b  = (const float*)d_in[14];
    const float* xproj_b_w = (const float*)d_in[15];
    const float* dtprj_b_w = (const float*)d_in[16];
    const float* dtprj_b_b = (const float*)d_in[17];
    const float* A_b_log   = (const float*)d_in[18];
    const float* D_b       = (const float*)d_in[19];
    const float* out_prj_w = (const float*)d_in[20];
    const float* dw1_w     = (const float*)d_in[21];
    const float* dw1_b     = (const float*)d_in[22];
    const float* pw1_w     = (const float*)d_in[23];
    const float* pw1_b     = (const float*)d_in[24];
    const float* dw2_w     = (const float*)d_in[25];
    const float* dw2_b     = (const float*)d_in[26];
    const float* pw2_w     = (const float*)d_in[27];
    const float* pw2_b     = (const float*)d_in[28];
    float* out = (float*)d_out;

    // ---- workspace arena (256B aligned) ----
    char* ws = (char*)d_ws;
    size_t off = 0;
    auto alloc = [&](size_t bytes) { size_t o = off; off += (bytes + 255) & ~(size_t)255; return o; };
    const size_t NT = NTOK;
    size_t o_xflat  = alloc(NT * DIMC * 4);
    size_t o_xn     = alloc(NT * DIMC * 2);
    size_t o_xz     = alloc(NT * 2 * DINNER * 4);
    size_t o_uf     = alloc(NT * DINNER * 4);
    size_t o_ub     = alloc(NT * DINNER * 4);
    size_t o_bcf    = alloc(NT * 38 * 4);
    size_t o_bcb    = alloc(NT * 38 * 4);
    size_t o_dlf    = alloc(NT * DINNER * 4);
    size_t o_dlb    = alloc(NT * DINNER * 4);
    size_t o_yf     = alloc(NT * DINNER * 4);
    size_t o_yb     = alloc(NT * DINNER * 4);
    size_t o_yc     = alloc(NT * DINNER * 2);
    size_t o_xm1    = alloc(NT * DIMC * 4);
    size_t o_winp   = alloc(2 * DINNER * DIMC * 2);
    size_t o_wout   = alloc(DIMC * DINNER * 2);
    size_t o_wpw1   = alloc(HIDC * DIMC * 2);
    size_t o_wpw2   = alloc(DIMC * HIDC * 2);
    // phase-2 overlays on freed scan buffers (each overlay <= region it reuses)
    size_t o_xn2 = o_uf;    // bf16  NT*96   (<= 25MB region)
    size_t o_g1  = o_ub;    // bf16  NT*96
    size_t o_h2  = o_dlf;   // bf16  NT*384  (25MB == region)
    size_t o_g2  = o_dlb;   // bf16  NT*384
    (void)ws_size; (void)in_sizes; (void)n_in; (void)out_size;

    float*  xflat = (float*)(ws + o_xflat);
    bf16_t* xn    = (bf16_t*)(ws + o_xn);
    float*  xz    = (float*)(ws + o_xz);
    float*  u_f   = (float*)(ws + o_uf);
    float*  u_b   = (float*)(ws + o_ub);
    float*  bc_f  = (float*)(ws + o_bcf);
    float*  bc_b  = (float*)(ws + o_bcb);
    float*  dl_f  = (float*)(ws + o_dlf);
    float*  dl_b  = (float*)(ws + o_dlb);
    float*  y_f   = (float*)(ws + o_yf);
    float*  y_b   = (float*)(ws + o_yb);
    bf16_t* yc    = (bf16_t*)(ws + o_yc);
    float*  xm1   = (float*)(ws + o_xm1);
    bf16_t* w_inp = (bf16_t*)(ws + o_winp);
    bf16_t* w_out = (bf16_t*)(ws + o_wout);
    bf16_t* w_p1  = (bf16_t*)(ws + o_wpw1);
    bf16_t* w_p2  = (bf16_t*)(ws + o_wpw2);
    bf16_t* xn2   = (bf16_t*)(ws + o_xn2);
    bf16_t* g1    = (bf16_t*)(ws + o_g1);
    bf16_t* h2    = (bf16_t*)(ws + o_h2);
    bf16_t* g2    = (bf16_t*)(ws + o_g2);

    // ---- weight conversion to bf16 ----
    f32_to_bf16_kernel<<<(2 * DINNER * DIMC + 255) / 256, 256, 0, stream>>>(in_proj_w, w_inp, 2 * DINNER * DIMC);
    f32_to_bf16_kernel<<<(DIMC * DINNER + 255) / 256, 256, 0, stream>>>(out_prj_w, w_out, DIMC * DINNER);
    f32_to_bf16_kernel<<<(HIDC * DIMC + 255) / 256, 256, 0, stream>>>(pw1_w, w_p1, HIDC * DIMC);
    f32_to_bf16_kernel<<<(DIMC * HIDC + 255) / 256, 256, 0, stream>>>(pw2_w, w_p2, DIMC * HIDC);

    // GEMM launch geometry: grid = M/16 row-tiles, blockDim = (N/16/3) waves.
    const int GRID_M  = NTOK / 16;          // 2048
    const int BLK_384 = (HIDC / 16 / 3) * 32;   // 8 waves = 256 threads
    const int BLK_96  = (DIMC / 16 / 3) * 32;   // 2 waves = 64 threads

    // ---- Mamba half ----
    ln1_prep_kernel<<<NTOK / 8, 256, 0, stream>>>(x_in, ln1_w, ln1_b, xflat, xn);
    // in_proj: (32768 x 96) @ (384 x 96)^T -> xz
    wmma_gemm_kernel<3><<<GRID_M, BLK_384, 0, stream>>>(
        xn, w_inp, nullptr, nullptr, xz, nullptr, NTOK, 2 * DINNER, DIMC, 0);
    conv_silu_kernel<<<NTOK * DINNER / 256, 256, 0, stream>>>(xz, conv_f_w, conv_f_b, u_f, 0);
    conv_silu_kernel<<<NTOK * DINNER / 256, 256, 0, stream>>>(xz, conv_b_w, conv_b_b, u_b, 1);
    xproj_kernel<<<NTOK * 38 / 256, 256, 0, stream>>>(u_f, xproj_f_w, bc_f);
    xproj_kernel<<<NTOK * 38 / 256, 256, 0, stream>>>(u_b, xproj_b_w, bc_b);
    delta_kernel<<<NTOK * DINNER / 256, 256, 0, stream>>>(bc_f, dtprj_f_w, dtprj_f_b, dl_f);
    delta_kernel<<<NTOK * DINNER / 256, 256, 0, stream>>>(bc_b, dtprj_b_w, dtprj_b_b, dl_b);
    scan_kernel<<<48, 256, 0, stream>>>(u_f, dl_f, bc_f, A_f_log, D_f, y_f);
    scan_kernel<<<48, 256, 0, stream>>>(u_b, dl_b, bc_b, A_b_log, D_b, y_b);
    gate_combine_kernel<<<NTOK * DINNER / 256, 256, 0, stream>>>(xz, y_f, y_b, yc);
    // out_proj + residual: (32768 x 192) @ (96 x 192)^T + xflat -> xm1
    wmma_gemm_kernel<3><<<GRID_M, BLK_96, 0, stream>>>(
        yc, w_out, nullptr, xflat, xm1, nullptr, NTOK, DIMC, DINNER, 0);

    // ---- MLP half ----
    ln2_kernel<<<NTOK / 8, 256, 0, stream>>>(xm1, ln2_w, ln2_b, xn2);
    dwconv3d_gelu_kernel<<<NTOK * DIMC / 256, 256, 0, stream>>>(xn2, dw1_w, dw1_b, g1, DIMC);
    wmma_gemm_kernel<3><<<GRID_M, BLK_384, 0, stream>>>(
        g1, w_p1, pw1_b, nullptr, nullptr, h2, NTOK, HIDC, DIMC, 1);
    dwconv3d_gelu_kernel<<<NTOK * HIDC / 256, 256, 0, stream>>>(h2, dw2_w, dw2_b, g2, HIDC);
    // pw2 + bias + residual, stored straight into (Bv*T, C, H, W) output layout
    wmma_gemm_kernel<3><<<GRID_M, BLK_96, 0, stream>>>(
        g2, w_p2, pw2_b, xm1, out, nullptr, NTOK, DIMC, HIDC, 2);
}
